// DynamicRoutingModel_84885733638835
// MI455X (gfx1250) — matrix operand order repaired and checked by reading, not compiled, not run on hardware
//
#include <hip/hip_runtime.h>
#include <cstddef>
#include <cstdint>

typedef __attribute__((ext_vector_type(16))) __bf16 v16bf;
typedef __attribute__((ext_vector_type(8)))  float  v8f;

// ---------------------------------------------------------------------------
// WMMA helpers (CDNA5 wave32, V_WMMA_F32_16X16X32_BF16)
// ---------------------------------------------------------------------------
__device__ inline v8f wmma_bf16(v16bf a, v16bf b, v8f c) {
  return __builtin_amdgcn_wmma_f32_16x16x32_bf16(
      /*neg_a=*/false, a, /*neg_b=*/false, b,
      /*c_mod=*/(short)0, c, /*reuse_a=*/false, /*reuse_b=*/false);
}

__device__ inline v8f v8f_zero() {
  v8f z;
#pragma unroll
  for (int i = 0; i < 8; ++i) z[i] = 0.f;
  return z;
}

// A-matrix 16x32 bf16 fragment: per-lane K granules of 8:
//   j<8  -> K = half*8 + j ;  j>=8 -> K = 16 + half*8 + (j-8)
__device__ inline v16bf frag_A(const __bf16* p, int half) {
  union { v16bf v; uint4 u[2]; } r;
  r.u[0] = *reinterpret_cast<const uint4*>(p + half * 8);
  r.u[1] = *reinterpret_cast<const uint4*>(p + 16 + half * 8);
  return r.v;
}

// B-matrix 32x16 bf16 fragment: per-lane K granules of 8:
//   j<8  -> K = half*8 + j ;  j>=8 -> K = 16 + half*8 + (j-8)
__device__ inline v16bf frag_B(const __bf16* p, int half) {
  union { v16bf v; uint4 u[2]; } r;
  r.u[0] = *reinterpret_cast<const uint4*>(p + half * 8);
  r.u[1] = *reinterpret_cast<const uint4*>(p + 16 + half * 8);
  return r.v;
}

// ---------------------------------------------------------------------------
// CDNA5 async global->LDS DMA (ASYNCcnt-tracked, no VGPR staging).
// INST_OFFSET is added to BOTH the LDS and global addresses (ISA ch.10),
// so one LDS-address VGPR + one global-address VGPR pair moves 64 bytes.
// LDS aperture keeps the wave-relative offset in the flat address' low 32
// bits, so truncating the generic pointer yields the DS address.
// ---------------------------------------------------------------------------
__device__ inline void async_copy64B(const __bf16* g, const __bf16* l) {
  const uint32_t la = (uint32_t)(uintptr_t)l;
  asm volatile(
      "global_load_async_to_lds_b128 %0, %1, off\n\t"
      "global_load_async_to_lds_b128 %0, %1, off offset:16\n\t"
      "global_load_async_to_lds_b128 %0, %1, off offset:32\n\t"
      "global_load_async_to_lds_b128 %0, %1, off offset:48"
      :
      : "v"(la), "v"(g)
      : "memory");
}

__device__ inline void wait_async0() {
  asm volatile("s_wait_asynccnt 0x0" ::: "memory");
}

// ---------------------------------------------------------------------------
// fp32 -> bf16 conversion (row-major preserved); n % 4 == 0
// ---------------------------------------------------------------------------
__global__ __launch_bounds__(256) void cvt_bf16_kernel(
    const float* __restrict__ x, __bf16* __restrict__ y, int n) {
  const int i = (blockIdx.x * 256 + threadIdx.x) * 4;
  if (i < n) {
    const float4 f = *reinterpret_cast<const float4*>(x + i);
    union { __bf16 h[4]; uint2 u; } pk;
    pk.h[0] = (__bf16)f.x;
    pk.h[1] = (__bf16)f.y;
    pk.h[2] = (__bf16)f.z;
    pk.h[3] = (__bf16)f.w;
    *reinterpret_cast<uint2*>(y + i) = pk.u;
  }
}

// fp32 B[K,N] -> bf16 BT[N,K] tiled transpose-convert (coalesced both sides)
__global__ __launch_bounds__(256) void cvt_transpose_kernel(
    const float* __restrict__ B, __bf16* __restrict__ BT, int K, int N) {
  __shared__ float t[32][33];
  const int k0 = blockIdx.x * 32, n0 = blockIdx.y * 32;
  const int tx = threadIdx.x & 31, ty = threadIdx.x >> 5;  // 8 rows/pass
#pragma unroll
  for (int r = ty; r < 32; r += 8) {
    const int n = n0 + tx;
    t[r][tx] = (n < N) ? B[(size_t)(k0 + r) * N + n] : 0.f;
  }
  __syncthreads();
#pragma unroll
  for (int r = ty; r < 32; r += 8) {
    const int n = n0 + r;
    if (n < N) BT[(size_t)n * K + k0 + tx] = (__bf16)t[tx][r];
  }
}

// ---------------------------------------------------------------------------
// GEMM: C[M,N] = A[M,K] @ B[K,N] with A bf16 row-major, BT bf16 N-major
// (pre-transposed), fp32 out. Block tile 128x128x64, double-buffered LDS
// filled by async global->LDS DMA, 256 threads = 8 waves, wave tile 32x64
// (16 WMMAs per K-stage). M % 128 == 0, K % 64 == 0; N guarded.
// ---------------------------------------------------------------------------
#define G_STRIDE 72  // 64 K-elems + 8 pad (bf16 units); 144B row, 16B aligned

__global__ __launch_bounds__(256) void gemm_bf16_kernel(
    const __bf16* __restrict__ A, const __bf16* __restrict__ BT,
    float* __restrict__ C, int M, int N, int K) {
  __shared__ __bf16 lA[2][128 * G_STRIDE];
  __shared__ __bf16 lB[2][128 * G_STRIDE];

  const int tid  = threadIdx.x;
  const int wave = tid >> 5;
  const int lane = tid & 31;
  const int hf   = lane >> 4;
  const int l16  = lane & 15;
  const int wm   = wave & 3;   // row group: wm*32
  const int wn   = wave >> 2;  // col group: wn*64
  const int bm   = blockIdx.x * 128;
  const int bn   = blockIdx.y * 128;

  // staging mapping: 2 threads per row, each a contiguous 64B strip
  const int srow = tid >> 1;
  const int sseg = (tid & 1) << 5;  // 0 / 32 (bf16 units)
  const __bf16* aptr = A + (size_t)(bm + srow) * K + sseg;
  const bool bok = (bn + srow) < N;
  const __bf16* bptr = BT + (size_t)(bn + srow) * K + sseg;
  __bf16* lap[2] = {&lA[0][srow * G_STRIDE + sseg],
                    &lA[1][srow * G_STRIDE + sseg]};
  __bf16* lbp[2] = {&lB[0][srow * G_STRIDE + sseg],
                    &lB[1][srow * G_STRIDE + sseg]};

  auto issue_stage = [&](int k0, int buf) {
    async_copy64B(aptr + k0, lap[buf]);
    if (bok) async_copy64B(bptr + k0, lbp[buf]);
  };

  v8f acc[2][4];
#pragma unroll
  for (int mt = 0; mt < 2; ++mt)
#pragma unroll
    for (int nt = 0; nt < 4; ++nt) acc[mt][nt] = v8f_zero();

  const int nk = K >> 6;
  issue_stage(0, 0);
  wait_async0();
  __syncthreads();

  for (int t = 0; t < nk; ++t) {
    const int cur = t & 1;
    if (t + 1 < nk) issue_stage((t + 1) << 6, 1 - cur);
    if (t + 2 < nk) {
      __builtin_prefetch(aptr + ((t + 2) << 6), 0, 1);
      if (bok) __builtin_prefetch(bptr + ((t + 2) << 6), 0, 1);
    }
#pragma unroll
    for (int ks = 0; ks < 64; ks += 32) {
      v16bf afr[2], bfr[4];
#pragma unroll
      for (int mt = 0; mt < 2; ++mt)
        afr[mt] =
            frag_A(&lA[cur][(wm * 32 + mt * 16 + l16) * G_STRIDE + ks], hf);
#pragma unroll
      for (int nt = 0; nt < 4; ++nt)
        bfr[nt] =
            frag_B(&lB[cur][(wn * 64 + nt * 16 + l16) * G_STRIDE + ks], hf);
#pragma unroll
      for (int mt = 0; mt < 2; ++mt)
#pragma unroll
        for (int nt = 0; nt < 4; ++nt)
          acc[mt][nt] = wmma_bf16(afr[mt], bfr[nt], acc[mt][nt]);
    }
    if (t + 1 < nk) wait_async0();
    __syncthreads();
  }

  // epilogue: C layout lane n=l16, VGPR r -> m = r + 8*half
#pragma unroll
  for (int mt = 0; mt < 2; ++mt)
#pragma unroll
    for (int nt = 0; nt < 4; ++nt) {
      const int col = bn + wn * 64 + nt * 16 + l16;
      if (col < N) {
        const int rbase = bm + wm * 32 + mt * 16 + hf * 8;
        float* cp = C + (size_t)rbase * N + col;
#pragma unroll
        for (int r = 0; r < 8; ++r) cp[(size_t)r * N] = acc[mt][nt][r];
      }
    }
}

// ---------------------------------------------------------------------------
// Flash attention with WMMA: grid (B*H, S/64), 128 threads (4 waves).
// qkv layout: [m][3][H=16][Dh=64] fp32 (m = b*S + s). O layout: [m][H*Dh].
// ---------------------------------------------------------------------------
#define A_STRIDE 72

__global__ __launch_bounds__(128) void flash_attn_kernel(
    const float* __restrict__ qkv, float* __restrict__ O, int Sseq) {
  __shared__ __bf16 sQ[64 * A_STRIDE];
  __shared__ __bf16 sK[64 * A_STRIDE];
  __shared__ __bf16 sV[64 * A_STRIDE];  // transposed: [d][token]
  __shared__ __bf16 sP[64 * A_STRIDE];

  const int bh  = blockIdx.x;
  const int b   = bh >> 4;
  const int h   = bh & 15;
  const int qt  = blockIdx.y;
  const int tid = threadIdx.x;
  const int w   = tid >> 5;
  const int lane = tid & 31;
  const int hf  = lane >> 4;
  const int l16 = lane & 15;

  for (int i = tid; i < 64 * 64; i += 128) {
    const int rr = i >> 6, dd = i & 63;
    const size_t m = (size_t)(b * Sseq + qt * 64 + rr);
    sQ[rr * A_STRIDE + dd] = (__bf16)qkv[m * 3072 + h * 64 + dd];
  }

  v8f oacc[4];
#pragma unroll
  for (int nt = 0; nt < 4; ++nt) oacc[nt] = v8f_zero();
  float rmax[8], rsum[8];
#pragma unroll
  for (int r = 0; r < 8; ++r) { rmax[r] = -3e38f; rsum[r] = 0.f; }

  for (int kt = 0; kt <= qt; ++kt) {
    __syncthreads();
    for (int i = tid; i < 64 * 64; i += 128) {
      const int rr = i >> 6, dd = i & 63;
      const size_t m = (size_t)(b * Sseq + kt * 64 + rr);
      sK[rr * A_STRIDE + dd] = (__bf16)qkv[m * 3072 + 1024 + h * 64 + dd];
      sV[dd * A_STRIDE + rr] = (__bf16)qkv[m * 3072 + 2048 + h * 64 + dd];
    }
    __syncthreads();

    v8f sacc[4];
#pragma unroll
    for (int nt = 0; nt < 4; ++nt) sacc[nt] = v8f_zero();
#pragma unroll
    for (int ks = 0; ks < 64; ks += 32) {
      const v16bf af = frag_A(&sQ[(w * 16 + l16) * A_STRIDE + ks], hf);
#pragma unroll
      for (int nt = 0; nt < 4; ++nt) {
        const v16bf bf = frag_B(&sK[(nt * 16 + l16) * A_STRIDE + ks], hf);
        sacc[nt] = wmma_bf16(af, bf, sacc[nt]);
      }
    }

    const float scl = 0.125f;  // 1/sqrt(64)
    if (kt == qt) {
#pragma unroll
      for (int nt = 0; nt < 4; ++nt)
#pragma unroll
        for (int r = 0; r < 8; ++r) {
          const int tk = nt * 16 + l16;
          const int qi = w * 16 + r + 8 * hf;
          sacc[nt][r] = (tk <= qi) ? sacc[nt][r] * scl : -3e38f;
        }
    } else {
#pragma unroll
      for (int nt = 0; nt < 4; ++nt)
#pragma unroll
        for (int r = 0; r < 8; ++r) sacc[nt][r] *= scl;
    }

    float nm[8], corr[8], ps[8];
#pragma unroll
    for (int r = 0; r < 8; ++r) {
      float mv = sacc[0][r];
#pragma unroll
      for (int nt = 1; nt < 4; ++nt) mv = fmaxf(mv, sacc[nt][r]);
#pragma unroll
      for (int off = 8; off > 0; off >>= 1)
        mv = fmaxf(mv, __shfl_xor(mv, off, 32));
      nm[r]   = fmaxf(rmax[r], mv);
      corr[r] = __expf(rmax[r] - nm[r]);
      ps[r]   = 0.f;
    }
#pragma unroll
    for (int nt = 0; nt < 4; ++nt)
#pragma unroll
      for (int r = 0; r < 8; ++r) {
        const float p = __expf(sacc[nt][r] - nm[r]);
        sacc[nt][r] = p;
        ps[r] += p;
      }
#pragma unroll
    for (int r = 0; r < 8; ++r) {
#pragma unroll
      for (int off = 8; off > 0; off >>= 1) ps[r] += __shfl_xor(ps[r], off, 32);
      rsum[r] = rsum[r] * corr[r] + ps[r];
      rmax[r] = nm[r];
    }
#pragma unroll
    for (int nt = 0; nt < 4; ++nt)
#pragma unroll
      for (int r = 0; r < 8; ++r) oacc[nt][r] *= corr[r];

#pragma unroll
    for (int nt = 0; nt < 4; ++nt)
#pragma unroll
      for (int r = 0; r < 8; ++r)
        sP[(w * 16 + r + 8 * hf) * A_STRIDE + nt * 16 + l16] =
            (__bf16)sacc[nt][r];

#pragma unroll
    for (int ks = 0; ks < 64; ks += 32) {
      const v16bf af = frag_A(&sP[(w * 16 + l16) * A_STRIDE + ks], hf);
#pragma unroll
      for (int nt = 0; nt < 4; ++nt) {
        const v16bf bf = frag_B(&sV[(nt * 16 + l16) * A_STRIDE + ks], hf);
        oacc[nt] = wmma_bf16(af, bf, oacc[nt]);
      }
    }
  }

#pragma unroll
  for (int nt = 0; nt < 4; ++nt)
#pragma unroll
    for (int r = 0; r < 8; ++r) {
      const int row = qt * 64 + w * 16 + r + 8 * hf;
      const size_t m = (size_t)(b * Sseq + row);
      O[m * 1024 + h * 64 + nt * 16 + l16] = oacc[nt][r] / rsum[r];
    }
}

// ---------------------------------------------------------------------------
// GDN delta-rule scan: one block per (b,h), 256 threads, 64x64 fp32 state.
// ---------------------------------------------------------------------------
__global__ __launch_bounds__(256) void gdn_scan_kernel(
    const float* __restrict__ Q, const float* __restrict__ K,
    const float* __restrict__ V, const float* __restrict__ beta,
    const float* __restrict__ decay, float* __restrict__ O, int Sseq) {
  const int bh = blockIdx.x;
  const int b = bh >> 4, h = bh & 15;
  const int tid = threadIdx.x;
  const int v = tid & 63;
  const int kseg = tid >> 6;

  float st[16];
#pragma unroll
  for (int i = 0; i < 16; ++i) st[i] = 0.f;

  __shared__ float kt[64], vt[64], qt[64], red[4][64], delta[64], sc[2];

  for (int t = 0; t < Sseq; ++t) {
    const size_t m = (size_t)b * Sseq + t;
    if (tid < 64)       kt[tid]       = K[m * 1024 + h * 64 + tid];
    else if (tid < 128) vt[tid - 64]  = V[m * 1024 + h * 64 + (tid - 64)];
    else if (tid < 192) qt[tid - 128] = Q[m * 1024 + h * 64 + (tid - 128)];
    else if (tid == 192) { sc[0] = beta[m * 16 + h]; sc[1] = decay[m * 16 + h]; }
    __syncthreads();

    const float dk = sc[1], bt = sc[0];
    float part = 0.f;
#pragma unroll
    for (int i = 0; i < 16; ++i) {
      st[i] *= dk;
      part += st[i] * kt[kseg * 16 + i];
    }
    red[kseg][v] = part;
    __syncthreads();
    if (kseg == 0) {
      const float vp = red[0][v] + red[1][v] + red[2][v] + red[3][v];
      delta[v] = (vt[v] - vp) * bt;
    }
    __syncthreads();
    const float dl = delta[v];
    float op = 0.f;
#pragma unroll
    for (int i = 0; i < 16; ++i) {
      st[i] += kt[kseg * 16 + i] * dl;
      op += st[i] * qt[kseg * 16 + i];
    }
    red[kseg][v] = op;
    __syncthreads();
    if (kseg == 0)
      O[m * 1024 + h * 64 + v] = red[0][v] + red[1][v] + red[2][v] + red[3][v];
    __syncthreads();
  }
}

// ---------------------------------------------------------------------------
// Elementwise / norm kernels
// ---------------------------------------------------------------------------
__global__ __launch_bounds__(256) void rmsnorm_kernel(
    const float* __restrict__ x, const float* __restrict__ w,
    float* __restrict__ y, int D) {
  const int row = blockIdx.x;
  const int tid = threadIdx.x;
  const float* xr = x + (size_t)row * D;
  float s = 0.f;
  for (int i = tid; i < D; i += 256) { const float v = xr[i]; s += v * v; }
#pragma unroll
  for (int off = 16; off > 0; off >>= 1) s += __shfl_xor(s, off, 32);
  __shared__ float r[8];
  if ((tid & 31) == 0) r[tid >> 5] = s;
  __syncthreads();
  float tot = 0.f;
#pragma unroll
  for (int i = 0; i < 8; ++i) tot += r[i];
  const float scale = rsqrtf(tot / (float)D + 1e-6f);
  float* yr = y + (size_t)row * D;
  for (int i = tid; i < D; i += 256) yr[i] = xr[i] * scale * w[i];
}

__global__ __launch_bounds__(64) void l2norm_kernel(float* __restrict__ x) {
  const int idx = blockIdx.x;
  const int d = threadIdx.x;
  float* p = x + (size_t)idx * 64;
  const float v = p[d];
  float s = v * v;
#pragma unroll
  for (int off = 16; off > 0; off >>= 1) s += __shfl_xor(s, off, 32);
  __shared__ float w2[2];
  if ((d & 31) == 0) w2[d >> 5] = s;
  __syncthreads();
  const float tot = w2[0] + w2[1];
  p[d] = v * rsqrtf(tot + 1e-6f);
}

__global__ __launch_bounds__(64) void outgate_kernel(
    float* __restrict__ o, const float* __restrict__ g,
    const float* __restrict__ gnorm) {
  const int idx = blockIdx.x;
  const int d = threadIdx.x;
  const size_t base = (size_t)idx * 64;
  const float v = o[base + d];
  float s = v * v;
#pragma unroll
  for (int off = 16; off > 0; off >>= 1) s += __shfl_xor(s, off, 32);
  __shared__ float w2[2];
  if ((d & 31) == 0) w2[d >> 5] = s;
  __syncthreads();
  const float mean = (w2[0] + w2[1]) / 64.f;
  const float gv = g[base + d];
  const float sig = gv / (1.f + __expf(-gv));
  o[base + d] = v * rsqrtf(mean + 1e-6f) * gnorm[d] * sig;
}

__global__ __launch_bounds__(256) void gate_kernel(
    const float* __restrict__ xb, const float* __restrict__ xa,
    float* __restrict__ beta, float* __restrict__ decay, int n) {
  const int i = blockIdx.x * 256 + threadIdx.x;
  if (i < n) {
    beta[i] = 1.f / (1.f + __expf(-xb[i]));
    const float a = xa[i];
    const float sp = (a > 20.f) ? a : log1pf(__expf(a));
    decay[i] = __expf(-sp);
  }
}

__global__ __launch_bounds__(256) void silu_mul_kernel(
    float* __restrict__ a, const float* __restrict__ u, int n) {
  const int i = blockIdx.x * 256 + threadIdx.x;
  if (i < n) {
    const float av = a[i];
    a[i] = av / (1.f + __expf(-av)) * u[i];
  }
}

__global__ __launch_bounds__(256) void add_kernel(
    float* __restrict__ dst, const float* __restrict__ src, int n) {
  const int i = blockIdx.x * 256 + threadIdx.x;
  if (i < n) dst[i] += src[i];
}

__global__ __launch_bounds__(256) void gather_kernel(
    const int* __restrict__ ids, const float* __restrict__ emb,
    float* __restrict__ x, int n, int D) {
  const int i = blockIdx.x * 256 + threadIdx.x;
  if (i < n) {
    const int m = i / D, d = i % D;
    x[i] = emb[(size_t)ids[m] * D + d];
  }
}

__global__ __launch_bounds__(256) void rope_kernel(float* __restrict__ qkv,
                                                   int Sseq, int n) {
  const int i = blockIdx.x * 256 + threadIdx.x;
  if (i >= n) return;
  const int m = i >> 9;
  const int r = i & 511;
  const int h = r >> 5;
  const int j = r & 31;
  const float freq = __powf(10000.f, -(float)(2 * j) / 64.f);
  const float ang = (float)(m % Sseq) * freq;
  const float c = __cosf(ang), sn = __sinf(ang);
  const size_t base = (size_t)m * 3072 + h * 64;
  const float q1 = qkv[base + j],        q2 = qkv[base + j + 32];
  const float k1 = qkv[base + 1024 + j], k2 = qkv[base + 1024 + j + 32];
  qkv[base + j]             = q1 * c - q2 * sn;
  qkv[base + j + 32]        = q2 * c + q1 * sn;
  qkv[base + 1024 + j]      = k1 * c - k2 * sn;
  qkv[base + 1024 + j + 32] = k2 * c + k1 * sn;
}

__global__ __launch_bounds__(256) void router_kernel(
    const float* __restrict__ rl, const float* __restrict__ rb,
    const float* __restrict__ u, float* __restrict__ r, int Mtot) {
  const int m = blockIdx.x * 256 + threadIdx.x;
  if (m >= Mtot) return;
  float a[4];
#pragma unroll
  for (int c = 0; c < 4; ++c) {
    const float noise = -__logf(-__logf(u[m * 4 + c] + 1e-10f) + 1e-10f);
    a[c] = rl[m * 4 + c] + rb[c] + noise;
  }
  const float s1 = (a[0] >= a[1]) ? 1.f : 0.f;
  const float s2 = (a[2] >= a[3]) ? 1.f : 0.f;
  r[m * 4 + 0] = s1;
  r[m * 4 + 1] = 1.f - s1;
  r[m * 4 + 2] = s2;
  r[m * 4 + 3] = 1.f - s2;
}

__global__ __launch_bounds__(256) void blend_kernel(
    const float* __restrict__ A, const float* __restrict__ Bv,
    const float* __restrict__ r, int off, float* __restrict__ x, int n,
    int D) {
  const int i = blockIdx.x * 256 + threadIdx.x;
  if (i < n) {
    const int m = i / D;
    x[i] = r[m * 4 + off] * A[i] + r[m * 4 + off + 1] * Bv[i];
  }
}

// ---------------------------------------------------------------------------
// Host orchestration
// ---------------------------------------------------------------------------
extern "C" void kernel_launch(void* const* d_in, const int* in_sizes, int n_in,
                              void* d_out, int out_size, void* d_ws,
                              size_t ws_size, hipStream_t stream) {
  (void)in_sizes; (void)out_size; (void)ws_size;
  const int Bb = 2, Ss = 1024, D = 1024, Ii = 4096;
  const int M = Bb * Ss;  // 2048

  const float* P[64];
  for (int i = 0; i < n_in && i < 64; ++i) P[i] = (const float*)d_in[i];
  const int* ids = (const int*)d_in[0];
  const float* gumbel = (const float*)d_in[1];
  const float* embed = P[2];
  const float* gdnp[3][13];
  for (int blk = 0; blk < 3; ++blk)
    for (int i = 0; i < 13; ++i) gdnp[blk][i] = P[3 + blk * 13 + i];

  // workspace layout
  float* w = (float*)d_ws;
  size_t off = 0;
  auto alloc = [&](size_t n) { float* p = w + off; off += n; return p; };
  float* X       = alloc((size_t)M * D);
  float* XN      = alloc((size_t)M * D);
  float* Qb      = alloc((size_t)M * D);
  float* Kb      = alloc((size_t)M * D);
  float* Vb      = alloc((size_t)M * D);
  float* Gb      = alloc((size_t)M * D);
  float* Obuf    = alloc((size_t)M * D);
  float* Y       = alloc((size_t)M * D);
  float* T1      = alloc((size_t)M * Ii);
  float* T2      = alloc((size_t)M * Ii);
  float* QKV     = alloc((size_t)M * 3 * D);
  float* GDNX    = alloc((size_t)M * D);
  float* ATTNOUT = alloc((size_t)M * D);
  float* BL      = alloc((size_t)M * 16);
  float* AL      = alloc((size_t)M * 16);
  float* BETA    = alloc((size_t)M * 16);
  float* DECAY   = alloc((size_t)M * 16);
  float* RL      = alloc((size_t)M * 4);
  float* Rr      = alloc((size_t)M * 4);
  __bf16* Abf = (__bf16*)alloc((size_t)M * Ii / 2);        // activations, bf16
  __bf16* Bbf = (__bf16*)alloc((size_t)32000 * 1024 / 2);  // weight^T, bf16

  auto blocks = [](int n) { return dim3((n + 255) / 256); };
  auto cvtA = [&](const float* src, int n) {
    cvt_bf16_kernel<<<(n / 4 + 255) / 256, 256, 0, stream>>>(src, Abf, n);
  };
  auto gemmW = [&](const float* W, float* Cc, int N_, int K_) {
    dim3 tg(K_ / 32, (N_ + 31) / 32);
    cvt_transpose_kernel<<<tg, 256, 0, stream>>>(W, Bbf, K_, N_);
    dim3 g(M / 128, (N_ + 127) / 128);
    gemm_bf16_kernel<<<g, 256, 0, stream>>>(Abf, Bbf, Cc, M, N_, K_);
  };

  auto gdn_block = [&](float* Xb, const float* const* g) {
    // g: n1,n2,Wq,Wk,Wv,Wb,Wa,Wg,g_norm,Wo,Wg_mlp,Wu,Wd
    rmsnorm_kernel<<<M, 256, 0, stream>>>(Xb, g[0], XN, D);
    cvtA(XN, M * D);
    gemmW(g[2], Qb, D, D);
    gemmW(g[3], Kb, D, D);
    gemmW(g[4], Vb, D, D);
    gemmW(g[7], Gb, D, D);
    gemmW(g[5], BL, 16, D);
    gemmW(g[6], AL, 16, D);
    l2norm_kernel<<<M * 16, 64, 0, stream>>>(Qb);
    l2norm_kernel<<<M * 16, 64, 0, stream>>>(Kb);
    gate_kernel<<<blocks(M * 16), 256, 0, stream>>>(BL, AL, BETA, DECAY, M * 16);
    gdn_scan_kernel<<<Bb * 16, 256, 0, stream>>>(Qb, Kb, Vb, BETA, DECAY, Obuf, Ss);
    outgate_kernel<<<M * 16, 64, 0, stream>>>(Obuf, Gb, g[8]);
    cvtA(Obuf, M * D);
    gemmW(g[9], Y, D, D);
    add_kernel<<<blocks(M * D), 256, 0, stream>>>(Xb, Y, M * D);
    rmsnorm_kernel<<<M, 256, 0, stream>>>(Xb, g[1], XN, D);
    cvtA(XN, M * D);
    gemmW(g[10], T1, Ii, D);
    gemmW(g[11], T2, Ii, D);
    silu_mul_kernel<<<blocks(M * Ii), 256, 0, stream>>>(T1, T2, M * Ii);
    cvtA(T1, M * Ii);
    gemmW(g[12], Y, D, Ii);
    add_kernel<<<blocks(M * D), 256, 0, stream>>>(Xb, Y, M * D);
  };

  auto attn_block = [&](const float* Xin, const float* Wqkv, const float* Wo,
                        float* Outb) {
    cvtA(Xin, M * D);
    gemmW(Wqkv, QKV, 3 * D, D);
    rope_kernel<<<blocks(M * 512), 256, 0, stream>>>(QKV, Ss, M * 512);
    flash_attn_kernel<<<dim3(Bb * 16, Ss / 64), 128, 0, stream>>>(QKV, Obuf, Ss);
    cvtA(Obuf, M * D);
    gemmW(Wo, Outb, D, D);
  };

  gather_kernel<<<blocks(M * D), 256, 0, stream>>>(ids, embed, X, M * D, D);
  gdn_block(X, gdnp[0]);
  // router on post-gdn0 x
  cvtA(X, M * D);
  gemmW(P[48], RL, 4, D);
  router_kernel<<<blocks(M), 256, 0, stream>>>(RL, P[49], gumbel, Rr, M);
  // layer 1
  hipMemcpyAsync(GDNX, X, (size_t)M * D * sizeof(float),
                 hipMemcpyDeviceToDevice, stream);
  attn_block(X, P[42], P[43], ATTNOUT);
  gdn_block(GDNX, gdnp[1]);
  blend_kernel<<<blocks(M * D), 256, 0, stream>>>(GDNX, ATTNOUT, Rr, 0, X,
                                                  M * D, D);
  // layer 2
  hipMemcpyAsync(GDNX, X, (size_t)M * D * sizeof(float),
                 hipMemcpyDeviceToDevice, stream);
  attn_block(X, P[44], P[45], ATTNOUT);
  gdn_block(GDNX, gdnp[2]);
  blend_kernel<<<blocks(M * D), 256, 0, stream>>>(GDNX, ATTNOUT, Rr, 2, X,
                                                  M * D, D);
  // attn3 (replaces x)
  attn_block(X, P[46], P[47], ATTNOUT);
  // final norm + LM head
  rmsnorm_kernel<<<M, 256, 0, stream>>>(ATTNOUT, P[50], XN, D);
  cvtA(XN, M * D);
  gemmW(P[51], (float*)d_out, 32000, D);
}
